// RGCNNet_50019189129231
// MI455X (gfx1250) — matrix operand order repaired and checked by reading, not compiled, not run on hardware
//
#include <hip/hip_runtime.h>
#include <math.h>

using bf16 = __bf16;
typedef __attribute__((ext_vector_type(16))) __bf16 v16bf;
typedef __attribute__((ext_vector_type(8)))  float  v8f;

union BfVec { v16bf v; uint4 u[2]; };

#define CDIV(a,b) (((a)+(b)-1)/(b))

// ---------------------------------------------------------------- utilities
__global__ void k_zero(uint4* __restrict__ p, size_t n4) {
    size_t i = (size_t)blockIdx.x * blockDim.x + threadIdx.x;
    if (i < n4) p[i] = make_uint4(0u, 0u, 0u, 0u);
}

__global__ void k_count(const int* __restrict__ dst, const int* __restrict__ et,
                        unsigned int* __restrict__ cnt, int N, int E) {
    int e = blockIdx.x * blockDim.x + threadIdx.x;
    if (e < E) atomicAdd(&cnt[(size_t)et[e] * N + dst[e]], 1u);
}

__global__ void k_inv(float* __restrict__ cnt, int n) {
    int i = blockIdx.x * blockDim.x + threadIdx.x;
    if (i >= n) return;
    int c = ((const int*)cnt)[i];
    float f = (float)c;
    cnt[i] = 1.0f / fmaxf(f, 1.0f);
}

// transpose-convert f32 [K][DOUT] -> bf16 [DOUT][K]
__global__ void k_cvt_t(const float* __restrict__ W, bf16* __restrict__ Wt, int K, int DOUT) {
    int i = blockIdx.x * blockDim.x + threadIdx.x;
    if (i >= K * DOUT) return;
    int k = i / DOUT, o = i - k * DOUT;
    Wt[(size_t)o * K + k] = (bf16)W[i];
}

__global__ void k_cvt(const float* __restrict__ s, bf16* __restrict__ d, size_t n) {
    size_t i = (size_t)blockIdx.x * blockDim.x + threadIdx.x;
    if (i < n) d[i] = (bf16)s[i];
}

__global__ void k_prep_txt(const float* __restrict__ txt, const float* __restrict__ mask,
                           bf16* __restrict__ out, int N) {
    size_t i = (size_t)blockIdx.x * blockDim.x + threadIdx.x;
    if (i >= (size_t)N * 384) return;
    int v = (int)(i / 384);
    out[i] = (bf16)(txt[i] * mask[v]);
}

// Wcat_t[DOUT][1152]: k<128 -> root[k][o]; else W_r[kk][o] = sum_b comp[r][b]*bases[b][kk][o]
__global__ void k_wcat(const float* __restrict__ comp, const float* __restrict__ bases,
                       const float* __restrict__ root, bf16* __restrict__ Wt, int DOUT) {
    int i = blockIdx.x * blockDim.x + threadIdx.x;
    if (i >= DOUT * 1152) return;
    int o = i % DOUT, k = i / DOUT;
    float val;
    if (k < 128) {
        val = root[(size_t)k * DOUT + o];
    } else {
        int r = (k - 128) >> 7, kk = (k - 128) & 127;
        val = 0.0f;
        #pragma unroll
        for (int b = 0; b < 8; ++b)
            val += comp[r * 8 + b] * bases[((size_t)b * 128 + kk) * DOUT + o];
    }
    Wt[(size_t)o * 1152 + k] = (bf16)val;
}

__global__ void k_combine(const float* __restrict__ num_x, const float* __restrict__ num_mask,
                          const float* __restrict__ W_num, const float* __restrict__ b_num,
                          const float* __restrict__ a_in, const float* __restrict__ b_txt,
                          const float* __restrict__ b_node,
                          const float* __restrict__ T1, const float* __restrict__ T2,
                          bf16* __restrict__ hout, int N) {
    size_t i = (size_t)blockIdx.x * blockDim.x + threadIdx.x;
    if (i >= (size_t)N * 128) return;
    int v = (int)(i >> 7), d = (int)(i & 127);
    float a = a_in[d];
    float pn = num_x[v] * num_mask[v] * W_num[d] + b_num[d];
    pn = pn >= 0.0f ? pn : a * pn;
    float pt = T1[i] + b_txt[d];
    pt = pt >= 0.0f ? pt : a * pt;
    float px = T2[i] + b_node[d];
    hout[i] = (bf16)(pn + pt + px);
}

// ------------------------------------------------- edge scatter (pk bf16 atomics)
__device__ inline unsigned int scale_pack(unsigned int u, float sc) {
    float lo = __uint_as_float(u << 16) * sc;
    float hi = __uint_as_float(u & 0xffff0000u) * sc;
    unsigned short slo = __builtin_bit_cast(unsigned short, (bf16)lo);
    unsigned short shi = __builtin_bit_cast(unsigned short, (bf16)hi);
    return (unsigned int)slo | ((unsigned int)shi << 16);
}

__device__ inline void atomic_pk_add_bf16(unsigned int* addr, unsigned int val) {
    asm volatile("global_atomic_pk_add_bf16 %0, %1, off" :: "v"(addr), "v"(val) : "memory");
}

__global__ __launch_bounds__(256) void k_scatter(const bf16* __restrict__ h,
                                                 const int* __restrict__ src,
                                                 const int* __restrict__ dst,
                                                 const int* __restrict__ et,
                                                 const float* __restrict__ inv,
                                                 bf16* __restrict__ Abuf, int N, int E) {
    int gid = blockIdx.x * blockDim.x + threadIdx.x;
    int e = gid >> 5;
    int lane = threadIdx.x & 31;
    if (e >= E) return;
    int s = src[e], d = dst[e], r = et[e];
    float sc = inv[(size_t)r * N + d];
    const unsigned int* hp = (const unsigned int*)(h + (size_t)s * 128) + lane * 2;
    unsigned int u0 = hp[0], u1 = hp[1];
    unsigned int* ap = (unsigned int*)(Abuf + ((size_t)r * N + d) * 128) + lane * 2;
    atomic_pk_add_bf16(ap,     scale_pack(u0, sc));
    atomic_pk_add_bf16(ap + 1, scale_pack(u1, sc));
}

// ------------------------------------------------------------- WMMA GEMM
// C[N,DOUT] = A[N,K](bf16) @ Wt[DOUT][K](bf16, pre-transposed)
// RGCN mode: A row v = concat(h[v], A_0[v], ..., A_7[v]) addressed as
//   Abase + (seg*N + v)*128 + kk, seg = k>>7.
// EPI: 0 = raw f32 store, 1 = bias+PReLU -> bf16, 2 = bias -> f32
template<int DOUT, int EPI, bool RGCN>
__global__ __launch_bounds__(256) void k_gemm(const bf16* __restrict__ Abase,
                                              const bf16* __restrict__ Wt,
                                              void* __restrict__ Cout,
                                              const float* __restrict__ bias,
                                              const float* __restrict__ alpha,
                                              int Nrows, int K) {
    constexpr int NT = DOUT / 16;
    const int lane = threadIdx.x & 31;
    const int wave = threadIdx.x >> 5;
    const int strip = blockIdx.x * 8 + wave;
    if (strip * 16 >= Nrows) return;          // wave-uniform
    const int lh = lane >> 4;
    const int lc = lane & 15;
    const int rowA = strip * 16 + lc;         // A: lane holds row M = lane&15
    const int kbA = lh ? 8 : 0;               // A K-base per lane half
    const int kbB = lh ? 8 : 0;               // B K-base per lane half

    v8f acc[NT];
    #pragma unroll
    for (int i = 0; i < NT; ++i)
        #pragma unroll
        for (int j = 0; j < 8; ++j) acc[i][j] = 0.0f;

    const int ksteps = K >> 5;
    for (int kt = 0; kt < ksteps; ++kt) {
        const bf16* pA;
        if constexpr (RGCN) {
            int seg = kt >> 2;
            int kk = ((kt & 3) << 5) + kbA;
            pA = Abase + ((size_t)seg * Nrows + rowA) * 128 + kk;
        } else {
            pA = Abase + (size_t)rowA * K + (kt << 5) + kbA;
        }
        BfVec a;
        a.u[0] = *reinterpret_cast<const uint4*>(pA);       // K = base .. base+7
        a.u[1] = *reinterpret_cast<const uint4*>(pA + 16);  // K = base+16 .. base+23
        const int kb = (kt << 5) + kbB;
        #pragma unroll
        for (int nt = 0; nt < NT; ++nt) {
            const bf16* pB = Wt + (size_t)(nt * 16 + lc) * K + kb;
            BfVec b;
            b.u[0] = *reinterpret_cast<const uint4*>(pB);
            b.u[1] = *reinterpret_cast<const uint4*>(pB + 16);
            acc[nt] = __builtin_amdgcn_wmma_f32_16x16x32_bf16(
                false, a.v, false, b.v, (short)0, acc[nt], false, false);
        }
    }

    #pragma unroll
    for (int nt = 0; nt < NT; ++nt) {
        #pragma unroll
        for (int j = 0; j < 8; ++j) {
            int r = strip * 16 + j + lh * 8;
            int c = nt * 16 + lc;
            float v = acc[nt][j];
            if constexpr (EPI == 0) {
                ((float*)Cout)[(size_t)r * DOUT + c] = v;
            } else if constexpr (EPI == 1) {
                v += bias[c];
                v = v >= 0.0f ? v : alpha[c] * v;
                ((bf16*)Cout)[(size_t)r * DOUT + c] = (bf16)v;
            } else {
                v += bias[c];
                ((float*)Cout)[(size_t)r * DOUT + c] = v;
            }
        }
    }
}

// ------------------------------------------------------------- log_softmax
__global__ void k_lsm(const float* __restrict__ logits, float* __restrict__ out, int N) {
    int v = blockIdx.x * blockDim.x + threadIdx.x;
    if (v >= N) return;
    float x[16];
    #pragma unroll
    for (int j = 0; j < 16; ++j) x[j] = logits[(size_t)v * 16 + j];
    float m = x[0];
    #pragma unroll
    for (int j = 1; j < 16; ++j) m = fmaxf(m, x[j]);
    float s = 0.0f;
    #pragma unroll
    for (int j = 0; j < 16; ++j) s += expf(x[j] - m);
    float l = logf(s) + m;
    #pragma unroll
    for (int j = 0; j < 16; ++j) out[(size_t)v * 16 + j] = x[j] - l;
}

// ------------------------------------------------------------------ launch
extern "C" void kernel_launch(void* const* d_in, const int* in_sizes, int n_in,
                              void* d_out, int out_size, void* d_ws, size_t ws_size,
                              hipStream_t stream) {
    const int N = in_sizes[0];      // num_x (N,1)
    const int E = in_sizes[6];      // edge_type (E,)
    constexpr int D = 128, TXT = 384, C = 16, R = 8, KC = 1152;

    const float* num_x    = (const float*)d_in[0];
    const float* num_mask = (const float*)d_in[1];
    const float* txt_x    = (const float*)d_in[2];
    const float* txt_mask = (const float*)d_in[3];
    const float* x        = (const float*)d_in[4];
    const int*   e_src    = (const int*)d_in[5];
    const int*   e_dst    = e_src + E;
    const int*   e_type   = (const int*)d_in[6];
    const float* W_num    = (const float*)d_in[7];
    const float* b_num    = (const float*)d_in[8];
    const float* W_txt    = (const float*)d_in[9];
    const float* b_txt    = (const float*)d_in[10];
    const float* W_node   = (const float*)d_in[11];
    const float* b_node   = (const float*)d_in[12];
    const float* a_in     = (const float*)d_in[13];
    const float* comps [3] = {(const float*)d_in[14], (const float*)d_in[19], (const float*)d_in[24]};
    const float* basesA[3] = {(const float*)d_in[15], (const float*)d_in[20], (const float*)d_in[25]};
    const float* roots [3] = {(const float*)d_in[16], (const float*)d_in[21], (const float*)d_in[26]};
    const float* biasA [3] = {(const float*)d_in[17], (const float*)d_in[22], (const float*)d_in[27]};
    const float* alphaA[2] = {(const float*)d_in[18], (const float*)d_in[23]};

    auto al = [](size_t o) { return (o + 255) & ~(size_t)255; };
    char* ws = (char*)d_ws;
    size_t off = 0;
    float* cntf   = (float*)(ws + off); off = al(off + (size_t)R * N * 4);
    bf16*  X      = (bf16*)(ws + off);  off = al(off + (size_t)9 * N * D * 2);
    char*  htmp   = ws + off;           off = al(off + (size_t)N * D * 2);
    bf16*  Wcat   = (bf16*)(ws + off);  off = al(off + (size_t)D * KC * 2);
    bf16*  WtxtT  = (bf16*)(ws + off);  off = al(off + (size_t)D * TXT * 2);
    bf16*  WnodeT = (bf16*)(ws + off);  off = al(off + (size_t)D * D * 2);

    bf16* hslot = X;
    bf16* Abuf  = X + (size_t)N * D;                 // [8][N][128] bf16 (N*2048 bytes)
    // embed scratch overlays the A region (dead before layer 1):
    bf16*  txtb = Abuf;                                            // N*768 B
    bf16*  xb   = (bf16*)((char*)Abuf + (size_t)N * 768);          // N*256 B
    float* T1f  = (float*)((char*)Abuf + (size_t)N * 1024);        // N*512 B
    float* T2f  = (float*)((char*)Abuf + (size_t)N * 1536);        // N*512 B

    const int strips  = CDIV(N, 16);
    const int gblocks = CDIV(strips, 8);

    // ---- edge counts -> inverse (shared by all 3 convs)
    k_zero <<<CDIV((size_t)2 * N, 256), 256, 0, stream>>>((uint4*)cntf, (size_t)2 * N);
    k_count<<<CDIV(E, 256), 256, 0, stream>>>(e_dst, e_type, (unsigned int*)cntf, N, E);
    k_inv  <<<CDIV(R * N, 256), 256, 0, stream>>>(cntf, R * N);

    // ---- input embedding
    k_cvt_t<<<CDIV(TXT * D, 256), 256, 0, stream>>>(W_txt, WtxtT, TXT, D);
    k_cvt_t<<<CDIV(D * D, 256), 256, 0, stream>>>(W_node, WnodeT, D, D);
    k_prep_txt<<<CDIV((size_t)N * TXT, 256), 256, 0, stream>>>(txt_x, txt_mask, txtb, N);
    k_gemm<128, 0, false><<<gblocks, 256, 0, stream>>>(txtb, WtxtT, T1f, nullptr, nullptr, N, TXT);
    k_cvt<<<CDIV((size_t)N * D, 256), 256, 0, stream>>>(x, xb, (size_t)N * D);
    k_gemm<128, 0, false><<<gblocks, 256, 0, stream>>>(xb, WnodeT, T2f, nullptr, nullptr, N, D);
    k_combine<<<CDIV((size_t)N * D, 256), 256, 0, stream>>>(
        num_x, num_mask, W_num, b_num, a_in, b_txt, b_node, T1f, T2f, hslot, N);

    // ---- conv1, conv2 (D -> D, PReLU)
    for (int L = 0; L < 2; ++L) {
        k_zero<<<CDIV((size_t)N * 128, 256), 256, 0, stream>>>((uint4*)Abuf, (size_t)N * 128);
        k_scatter<<<CDIV((size_t)E * 32, 256), 256, 0, stream>>>(
            hslot, e_src, e_dst, e_type, cntf, Abuf, N, E);
        k_wcat<<<CDIV(D * KC, 256), 256, 0, stream>>>(comps[L], basesA[L], roots[L], Wcat, D);
        k_gemm<128, 1, true><<<gblocks, 256, 0, stream>>>(
            X, Wcat, htmp, biasA[L], alphaA[L], N, KC);
        hipMemcpyAsync(hslot, htmp, (size_t)N * D * sizeof(bf16),
                       hipMemcpyDeviceToDevice, stream);
    }

    // ---- conv3 (D -> 16) + log_softmax
    k_zero<<<CDIV((size_t)N * 128, 256), 256, 0, stream>>>((uint4*)Abuf, (size_t)N * 128);
    k_scatter<<<CDIV((size_t)E * 32, 256), 256, 0, stream>>>(
        hslot, e_src, e_dst, e_type, cntf, Abuf, N, E);
    k_wcat<<<CDIV(C * KC, 256), 256, 0, stream>>>(comps[2], basesA[2], roots[2], Wcat, C);
    k_gemm<16, 2, true><<<gblocks, 256, 0, stream>>>(
        X, Wcat, htmp, biasA[2], nullptr, N, KC);
    k_lsm<<<CDIV(N, 256), 256, 0, stream>>>((const float*)htmp, (float*)d_out, N);
}
